// DRASIEncoder_47579647705647
// MI455X (gfx1250) — matrix operand order repaired and checked by reading, not compiled, not run on hardware
//
#include <hip/hip_runtime.h>

#define NNODES 50000
#define NEDGES 600000
#define IN_DIM 512
#define HID 128
#define LAT 32

typedef __attribute__((ext_vector_type(16))) _Float16 v16h;
typedef __attribute__((ext_vector_type(8)))  _Float16 v8h;
typedef __attribute__((ext_vector_type(4)))  _Float16 v4h;
typedef __attribute__((ext_vector_type(8)))  float    v8f;

// ---------------------------------------------------------------------------
// A fragment (16x32 f16, CDNA5 ISA §7.12.2), lane=(g=lane>>4,
// mr=lane&15): elems[0..7]=A[row,k0+8g+0..7], elems[8..15]=A[row,k0+8g+16..23].
// Caller passes p = &A[row*K + k0 + 8*g].  f32 source: convert in-register.
__device__ __forceinline__ v16h load_a_frag(const float* __restrict__ p) {
  float4 f0 = *reinterpret_cast<const float4*>(p + 0);
  float4 f1 = *reinterpret_cast<const float4*>(p + 4);
  float4 f2 = *reinterpret_cast<const float4*>(p + 16);
  float4 f3 = *reinterpret_cast<const float4*>(p + 20);
  v16h a;
  a[0]=(_Float16)f0.x; a[1]=(_Float16)f0.y; a[2]=(_Float16)f0.z; a[3]=(_Float16)f0.w;
  a[4]=(_Float16)f1.x; a[5]=(_Float16)f1.y; a[6]=(_Float16)f1.z; a[7]=(_Float16)f1.w;
  a[8]=(_Float16)f2.x; a[9]=(_Float16)f2.y; a[10]=(_Float16)f2.z; a[11]=(_Float16)f2.w;
  a[12]=(_Float16)f3.x; a[13]=(_Float16)f3.y; a[14]=(_Float16)f3.z; a[15]=(_Float16)f3.w;
  return a;
}
// f16 source: two b128 loads, no VALU.
__device__ __forceinline__ v16h load_a_frag(const _Float16* __restrict__ p) {
  v8h lo = *reinterpret_cast<const v8h*>(p);
  v8h hi = *reinterpret_cast<const v8h*>(p + 16);
  v16h a;
#pragma unroll
  for (int i = 0; i < 8; ++i) { a[i] = lo[i]; a[8 + i] = hi[i]; }
  return a;
}
// B fragment (32x16; B[k,n]=W[n,k], W row-major f16 [N,K]):
// elems[0..7] = W[col, k0+8g+0..7], elems[8..15] = W[col, k0+8g+16..23].
__device__ __forceinline__ v16h load_b_frag(const _Float16* __restrict__ p) {
  v8h lo = *reinterpret_cast<const v8h*>(p);
  v8h hi = *reinterpret_cast<const v8h*>(p + 16);
  v16h b;
#pragma unroll
  for (int i = 0; i < 8; ++i) { b[i] = lo[i]; b[8 + i] = hi[i]; }
  return b;
}

// ---------------------------------------------------------------------------
// out_f16[M,128] = act( A[M,K1] @ W1[128,K1]^T (+ A2[M,128] @ W2[128,128]^T)
//                       + bias ).  One wave = one 16x128 tile, 8 waves/block.
template <typename TA1>
__global__ __launch_bounds__(256)
void gemm_wmma_n128(const TA1* __restrict__ A, int K1,
                    const _Float16* __restrict__ Wh1,
                    const _Float16* __restrict__ A2,   // optional, K=HID
                    const _Float16* __restrict__ Wh2,
                    const float* __restrict__ bias,
                    _Float16* __restrict__ out, int M, int relu)
{
  const int wave = threadIdx.x >> 5;
  const int lane = threadIdx.x & 31;
  const int m0 = (blockIdx.x * 8 + wave) * 16;
  if (m0 >= M) return;
  const int g  = lane >> 4;
  const int mr = lane & 15;

  v8f c[8];
#pragma unroll
  for (int i = 0; i < 8; ++i) c[i] = (v8f){0.f,0.f,0.f,0.f,0.f,0.f,0.f,0.f};

  {
    const TA1* arow = A + (size_t)(m0 + mr) * K1 + 8 * g;
    for (int k0 = 0; k0 < K1; k0 += 32) {
      v16h a = load_a_frag(arow + k0);
#pragma unroll
      for (int nt = 0; nt < 8; ++nt) {
        v16h b = load_b_frag(Wh1 + (size_t)(nt * 16 + mr) * K1 + k0 + 8 * g);
        c[nt] = __builtin_amdgcn_wmma_f32_16x16x32_f16(
            false, a, false, b, (short)0, c[nt], false, false);
      }
    }
  }
  if (A2 != nullptr) {
    const _Float16* arow = A2 + (size_t)(m0 + mr) * HID + 8 * g;
    for (int k0 = 0; k0 < HID; k0 += 32) {
      v16h a = load_a_frag(arow + k0);
#pragma unroll
      for (int nt = 0; nt < 8; ++nt) {
        v16h b = load_b_frag(Wh2 + (size_t)(nt * 16 + mr) * HID + k0 + 8 * g);
        c[nt] = __builtin_amdgcn_wmma_f32_16x16x32_f16(
            false, a, false, b, (short)0, c[nt], false, false);
      }
    }
  }

  // C/D layout: VGPR r holds D[m0 + r + 8g, nt*16 + mr].
#pragma unroll
  for (int nt = 0; nt < 8; ++nt) {
    const int n = nt * 16 + mr;
    const float bv = bias ? bias[n] : 0.f;
#pragma unroll
    for (int r = 0; r < 8; ++r) {
      const int m = m0 + r + 8 * g;
      float v = c[nt][r] + bv;
      if (relu) v = v > 0.f ? v : 0.f;
      out[(size_t)m * HID + n] = (_Float16)v;
    }
  }
}

// ---------------------------------------------------------------------------
// Heads: mu = H @ muW^T + mub ; lv = H @ lvW^T + lvb ; d_out = [mu | lv] f32.
__global__ __launch_bounds__(256)
void heads_wmma(const _Float16* __restrict__ H,
                const _Float16* __restrict__ muW, const float* __restrict__ mub,
                const _Float16* __restrict__ lvW, const float* __restrict__ lvb,
                float* __restrict__ out, int M)
{
  const int wave = threadIdx.x >> 5;
  const int lane = threadIdx.x & 31;
  const int m0 = (blockIdx.x * 8 + wave) * 16;
  if (m0 >= M) return;
  const int g  = lane >> 4;
  const int mr = lane & 15;

  v8f c[4];
#pragma unroll
  for (int i = 0; i < 4; ++i) c[i] = (v8f){0.f,0.f,0.f,0.f,0.f,0.f,0.f,0.f};

  const _Float16* arow = H + (size_t)(m0 + mr) * HID + 8 * g;
  for (int k0 = 0; k0 < HID; k0 += 32) {
    v16h a = load_a_frag(arow + k0);
#pragma unroll
    for (int t = 0; t < 4; ++t) {
      const _Float16* W = (t < 2) ? muW : lvW;
      const int col = (t & 1) * 16 + mr;
      v16h b = load_b_frag(W + (size_t)col * HID + k0 + 8 * g);
      c[t] = __builtin_amdgcn_wmma_f32_16x16x32_f16(
          false, a, false, b, (short)0, c[t], false, false);
    }
  }

#pragma unroll
  for (int t = 0; t < 4; ++t) {
    const int n = (t & 1) * 16 + mr;
    const float bv = (t < 2) ? mub[n] : lvb[n];
    float* op = out + ((t < 2) ? (size_t)0 : (size_t)M * LAT);
#pragma unroll
    for (int r = 0; r < 8; ++r) {
      const int m = m0 + r + 8 * g;
      op[(size_t)m * LAT + n] = c[t][r] + bv;
    }
  }
}

// ---------------------------------------------------------------------------
// agg[dst] += ew * h[src]; one wave/edge, 4 f16 channels per lane (b64 gather),
// f32 global atomics preserve segment-sum precision.
__global__ __launch_bounds__(256)
void edge_scatter(const _Float16* __restrict__ h,
                  const int* __restrict__ src, const int* __restrict__ dst,
                  const float* __restrict__ ew,
                  float* __restrict__ agg, int E)
{
  const int t = blockIdx.x * 256 + (int)threadIdx.x;
  const int e = t >> 5;
  const int lane = t & 31;
  if (e >= E) return;
  const int s = src[e];
  const int d = dst[e];
  const float w = ew[e];
  v4h hv = *reinterpret_cast<const v4h*>(h + (size_t)s * HID + lane * 4);
  float* ap = agg + (size_t)d * HID + lane * 4;
  atomicAdd(ap + 0, (float)hv[0] * w);
  atomicAdd(ap + 1, (float)hv[1] * w);
  atomicAdd(ap + 2, (float)hv[2] * w);
  atomicAdd(ap + 3, (float)hv[3] * w);
}

__global__ __launch_bounds__(256)
void zero_f32v4(float4* __restrict__ p, int n4)
{
  const int i = blockIdx.x * 256 + (int)threadIdx.x;
  if (i < n4) p[i] = make_float4(0.f, 0.f, 0.f, 0.f);
}

// One-shot weight conversion f32 -> f16 (4-wide).
__global__ __launch_bounds__(256)
void cvt_f32_f16v4(const float* __restrict__ s, _Float16* __restrict__ d, int n4)
{
  const int i = blockIdx.x * 256 + (int)threadIdx.x;
  if (i < n4) {
    float4 f = reinterpret_cast<const float4*>(s)[i];
    v4h hv;
    hv[0] = (_Float16)f.x; hv[1] = (_Float16)f.y;
    hv[2] = (_Float16)f.z; hv[3] = (_Float16)f.w;
    reinterpret_cast<v4h*>(d)[i] = hv;
  }
}

// ---------------------------------------------------------------------------
extern "C" void kernel_launch(void* const* d_in, const int* in_sizes, int n_in,
                              void* d_out, int out_size, void* d_ws, size_t ws_size,
                              hipStream_t stream) {
  const float* x    = (const float*)d_in[0];
  const int*   ei   = (const int*)d_in[1];     // [2, E]
  const float* ew   = (const float*)d_in[2];
  const float* W1   = (const float*)d_in[3];
  const float* b1   = (const float*)d_in[4];
  const float* W2   = (const float*)d_in[5];
  const float* b2   = (const float*)d_in[6];
  const float* g1rW = (const float*)d_in[7];
  const float* g1rb = (const float*)d_in[8];
  const float* g1oW = (const float*)d_in[9];
  const float* g2rW = (const float*)d_in[10];
  const float* g2rb = (const float*)d_in[11];
  const float* g2oW = (const float*)d_in[12];
  const float* muW  = (const float*)d_in[13];
  const float* mub  = (const float*)d_in[14];
  const float* lvW  = (const float*)d_in[15];
  const float* lvb  = (const float*)d_in[16];
  float* out = (float*)d_out;

  const int M = NNODES;
  const size_t NH = (size_t)M * HID;

  // Workspace: H0,H1 f16 activations; AGG f32; f16 weight copies.
  _Float16* H0  = (_Float16*)d_ws;        // h1 / h3
  _Float16* H1  = H0 + NH;                // h2 / h4
  float*    AGG = (float*)(H1 + NH);      // zeroed each call
  _Float16* W1h   = (_Float16*)(AGG + NH);
  _Float16* W2h   = W1h + (size_t)HID * IN_DIM;
  _Float16* g1rWh = W2h   + (size_t)HID * HID;
  _Float16* g1oWh = g1rWh + (size_t)HID * HID;
  _Float16* g2rWh = g1oWh + (size_t)HID * HID;
  _Float16* g2oWh = g2rWh + (size_t)HID * HID;
  _Float16* muWh  = g2oWh + (size_t)HID * HID;
  _Float16* lvWh  = muWh  + (size_t)LAT * HID;

  const int* src = ei;
  const int* dst = ei + NEDGES;

  dim3 blk(256);
  const int gemm_blocks = (M / 16 + 7) / 8;            // 391
  const int zero_blocks = (int)((NH / 4 + 255) / 256); // 6250
  const int edge_blocks = (NEDGES * 32) / 256;         // 75000
  auto cvtb = [](int n) { return (n / 4 + 255) / 256; };

  // Pre-convert weights (once per call; ~300 KB total).
  cvt_f32_f16v4<<<cvtb(HID*IN_DIM), blk, 0, stream>>>(W1,   W1h,   HID*IN_DIM/4);
  cvt_f32_f16v4<<<cvtb(HID*HID),    blk, 0, stream>>>(W2,   W2h,   HID*HID/4);
  cvt_f32_f16v4<<<cvtb(HID*HID),    blk, 0, stream>>>(g1rW, g1rWh, HID*HID/4);
  cvt_f32_f16v4<<<cvtb(HID*HID),    blk, 0, stream>>>(g1oW, g1oWh, HID*HID/4);
  cvt_f32_f16v4<<<cvtb(HID*HID),    blk, 0, stream>>>(g2rW, g2rWh, HID*HID/4);
  cvt_f32_f16v4<<<cvtb(HID*HID),    blk, 0, stream>>>(g2oW, g2oWh, HID*HID/4);
  cvt_f32_f16v4<<<cvtb(LAT*HID),    blk, 0, stream>>>(muW,  muWh,  LAT*HID/4);
  cvt_f32_f16v4<<<cvtb(LAT*HID),    blk, 0, stream>>>(lvW,  lvWh,  LAT*HID/4);

  // h1 = relu(x @ W1^T + b1)
  gemm_wmma_n128<float><<<gemm_blocks, blk, 0, stream>>>(
      x, IN_DIM, W1h, nullptr, nullptr, b1, H0, M, 1);
  // h2 = relu(h1 @ W2^T + b2)
  gemm_wmma_n128<_Float16><<<gemm_blocks, blk, 0, stream>>>(
      H0, HID, W2h, nullptr, nullptr, b2, H1, M, 1);
  // conv1: agg = segment_sum(ew * h2[src] -> dst)
  zero_f32v4<<<zero_blocks, blk, 0, stream>>>((float4*)AGG, (int)(NH / 4));
  edge_scatter<<<edge_blocks, blk, 0, stream>>>(H1, src, dst, ew, AGG, NEDGES);
  // h3 = relu(agg @ relW^T + relb + h2 @ rootW^T)
  gemm_wmma_n128<float><<<gemm_blocks, blk, 0, stream>>>(
      AGG, HID, g1rWh, H1, g1oWh, g1rb, H0, M, 1);
  // conv2
  zero_f32v4<<<zero_blocks, blk, 0, stream>>>((float4*)AGG, (int)(NH / 4));
  edge_scatter<<<edge_blocks, blk, 0, stream>>>(H0, src, dst, ew, AGG, NEDGES);
  gemm_wmma_n128<float><<<gemm_blocks, blk, 0, stream>>>(
      AGG, HID, g2rWh, H0, g2oWh, g2rb, H1, M, 1);
  // heads -> d_out = [mu | logvar]
  heads_wmma<<<gemm_blocks, blk, 0, stream>>>(H1, muWh, mub, lvWh, lvb, out, M);
  (void)in_sizes; (void)n_in; (void)out_size; (void)ws_size;
}
